// CausalSelfAttention_6605659701452
// MI455X (gfx1250) — 16-bit and 8-bit matrix operand order repaired and checked by reading, compiled, not run on hardware
//
#include <hip/hip_runtime.h>
#include <hip/hip_bf16.h>
#include <stdint.h>

typedef __attribute__((ext_vector_type(16))) _Float16 v16h;
typedef __attribute__((ext_vector_type(8)))  float    v8f;
typedef __attribute__((ext_vector_type(8)))  int      v8i;

#define BDIM  2
#define TSEQ  2048
#define CDIM  1024
#define NHEAD 16
#define HD    64
#define EPSF  1.1920929e-07f
#define SCALEF 0.12f

union UH16 { v16h v; _Float16 h[16]; };
union UF8  { v8f  v; float    f[8];  };
union UI8  { v8i  v; int      i[8];  };

typedef int v4i_vec __attribute__((vector_size(16)));

__device__ __forceinline__ void async_copy16(const void* g, void* l) {
#if __has_builtin(__builtin_amdgcn_global_load_async_to_lds_b128)
  __builtin_amdgcn_global_load_async_to_lds_b128(
      (__attribute__((address_space(1))) v4i_vec*)const_cast<void*>(g),
      (__attribute__((address_space(3))) v4i_vec*)l, 0, 0);
#else
  *(int4*)l = *(const int4*)g;
#endif
}
__device__ __forceinline__ void wait_async0() {
  asm volatile("s_wait_asynccnt 0x0" ::: "memory");
}

__global__ __launch_bounds__(256) void k_abs_partial(const float* __restrict__ w, int n,
                                                     float* __restrict__ part) {
  __shared__ float red[256];
  int tid = threadIdx.x;
  float s = 0.f;
  for (int i = blockIdx.x * 256 + tid; i < n; i += gridDim.x * 256) s += fabsf(w[i]);
  red[tid] = s; __syncthreads();
  for (int st = 128; st > 0; st >>= 1) { if (tid < st) red[tid] += red[tid + st]; __syncthreads(); }
  if (tid == 0) part[blockIdx.x] = red[0];
}

__global__ __launch_bounds__(256) void k_abs_final(const float* __restrict__ part, int np,
                                                   float invn, float* __restrict__ scal) {
  __shared__ float red[256];
  int tid = threadIdx.x;
  red[tid] = (tid < np) ? part[tid] : 0.f; __syncthreads();
  for (int st = 128; st > 0; st >>= 1) { if (tid < st) red[tid] += red[tid + st]; __syncthreads(); }
  if (tid == 0) {
    float mean = fmaxf(red[0] * invn, 1e-5f);
    scal[0] = 1.f / mean;
    scal[1] = mean;
  }
}

__global__ __launch_bounds__(256) void k_quant_w(const float* __restrict__ w,
                                                 const float* __restrict__ scal,
                                                 signed char* __restrict__ wq, int n) {
  int i = blockIdx.x * 256 + threadIdx.x;
  if (i >= n) return;
  float ws = scal[0];
  float q = fminf(fmaxf(rintf(w[i] * ws), -1.f), 1.f);
  wq[i] = (signed char)(int)q;
}

__global__ __launch_bounds__(256) void k_quant_act(const float* __restrict__ X,
                                                   signed char* __restrict__ Xq,
                                                   float* __restrict__ invA) {
  __shared__ float rs_[256], rm_[256];
  int tid = threadIdx.x;
  const float* row = X + (size_t)blockIdx.x * CDIM;
  float vals[4]; float ss = 0.f, mx = 0.f;
#pragma unroll
  for (int i = 0; i < 4; ++i) {
    float v = row[tid + i * 256];
    vals[i] = v; ss += v * v; mx = fmaxf(mx, fabsf(v));
  }
  rs_[tid] = ss; rm_[tid] = mx; __syncthreads();
  for (int st = 128; st > 0; st >>= 1) {
    if (tid < st) { rs_[tid] += rs_[tid + st]; rm_[tid] = fmaxf(rm_[tid], rm_[tid + st]); }
    __syncthreads();
  }
  float rnorm = rsqrtf(rs_[0] * (1.f / CDIM) + EPSF);
  float nmax  = fmaxf(rm_[0] * rnorm, 1e-5f);
  float ascal = 127.f / nmax;
  if (tid == 0) invA[blockIdx.x] = nmax * (1.f / 127.f);
  signed char* orow = Xq + (size_t)blockIdx.x * CDIM;
#pragma unroll
  for (int i = 0; i < 4; ++i) {
    float q = fminf(fmaxf(rintf(vals[i] * rnorm * ascal), -128.f), 127.f);
    orow[tid + i * 256] = (signed char)(int)q;
  }
}

__global__ __launch_bounds__(256) void k_gemm_iu8(const signed char* __restrict__ Aq,
                                                  const signed char* __restrict__ Bq,
                                                  const float* __restrict__ invA,
                                                  const float* __restrict__ scal,
                                                  float* __restrict__ C, int N) {
  __shared__ __align__(16) signed char lA[64 * 64];
  __shared__ __align__(16) signed char lB[128 * 64];
  const int tid = threadIdx.x, wid = tid >> 5, lane = tid & 31;
  const int hi = lane >> 4, ln = lane & 15;
  const int wm = wid >> 2, wn = wid & 3;
  const int bm = blockIdx.y * 64, bn = blockIdx.x * 128;
  const float invW = scal[1];
  UI8 acc[2][2];
#pragma unroll
  for (int a = 0; a < 2; ++a)
#pragma unroll
    for (int b = 0; b < 2; ++b)
#pragma unroll
      for (int j = 0; j < 8; ++j) acc[a][b].i[j] = 0;

  const int ar = tid >> 2, ac = (tid & 3) * 16;
  for (int kt = 0; kt < CDIM; kt += 64) {
    async_copy16(Aq + (size_t)(bm + ar) * CDIM + kt + ac,      lA + ar * 64 + ac);
    async_copy16(Bq + (size_t)(bn + ar) * CDIM + kt + ac,      lB + ar * 64 + ac);
    async_copy16(Bq + (size_t)(bn + ar + 64) * CDIM + kt + ac, lB + (ar + 64) * 64 + ac);
    wait_async0();
    __syncthreads();
    UI8 af[2], bf[2];
#pragma unroll
    for (int mt = 0; mt < 2; ++mt) {
      int row = wm * 32 + mt * 16 + ln;
#pragma unroll
      for (int j = 0; j < 8; ++j) {
        int kb = (j >> 1) * 16 + (j & 1) * 4 + hi * 8;
        af[mt].i[j] = *(const int*)(lA + row * 64 + kb);
      }
    }
#pragma unroll
    for (int nt = 0; nt < 2; ++nt) {
      int col = wn * 32 + nt * 16 + ln;
#pragma unroll
      for (int j = 0; j < 8; ++j) {
        int kb = (j >> 1) * 16 + (j & 1) * 4 + hi * 8;
        bf[nt].i[j] = *(const int*)(lB + col * 64 + kb);
      }
    }
#pragma unroll
    for (int mt = 0; mt < 2; ++mt)
#pragma unroll
      for (int nt = 0; nt < 2; ++nt)
        acc[mt][nt].v = __builtin_amdgcn_wmma_i32_16x16x64_iu8(
            true, af[mt].v, true, bf[nt].v, acc[mt][nt].v, false, false);
    __syncthreads();
  }
#pragma unroll
  for (int mt = 0; mt < 2; ++mt)
#pragma unroll
    for (int j = 0; j < 8; ++j) {
      int rg = bm + wm * 32 + mt * 16 + hi * 8 + j;
      float d = invA[rg] * invW;
#pragma unroll
      for (int nt = 0; nt < 2; ++nt) {
        int cg = bn + wn * 32 + nt * 16 + ln;
        C[(size_t)rg * N + cg] = (float)acc[mt][nt].i[j] * d;
      }
    }
}

__global__ __launch_bounds__(256) void k_qkv_post(const float* __restrict__ qkv,
                                                  _Float16* __restrict__ Qh,
                                                  _Float16* __restrict__ Kh,
                                                  _Float16* __restrict__ Vt) {
  int wid = threadIdx.x >> 5, lane = threadIdx.x & 31;
  int r  = blockIdx.x * 8 + wid;
  int sel = r % 3; int r2 = r / 3;
  int t  = r2 & (TSEQ - 1);
  int bh = r2 >> 11;
  int b = bh >> 4, h = bh & 15;
  const float* src = qkv + ((size_t)(b * TSEQ + t)) * (3 * CDIM) + sel * CDIM + h * HD;
  float x0 = src[lane], x1 = src[lane + 32];
  float ss = x0 * x0 + x1 * x1;
#pragma unroll
  for (int o = 16; o > 0; o >>= 1) ss += __shfl_xor(ss, o, 32);
  float rn = rsqrtf(ss * (1.f / HD) + EPSF);
  float n0 = x0 * rn, n1 = x1 * rn;
  if (sel == 2) {
    Vt[((size_t)bh * HD + lane)      * TSEQ + t] = (_Float16)n0;
    Vt[((size_t)bh * HD + lane + 32) * TSEQ + t] = (_Float16)n1;
    return;
  }
  float af = (lane < 16) ? exp2f(-10.f * (float)lane * (1.f / 15.f)) : 0.f;
  float th = (float)h * af;
  float c = cosf(th), s = sinf(th);
  float o0 =  n0 * c + n1 * s;
  float o1 = -n0 * s + n1 * c;
  _Float16* dst = (sel == 0 ? Qh : Kh) + (size_t)r2 * HD;
  dst[lane] = (_Float16)o0; dst[lane + 32] = (_Float16)o1;
}

__global__ __launch_bounds__(256) void k_attn(const _Float16* __restrict__ Qh,
                                              const _Float16* __restrict__ Kh,
                                              const _Float16* __restrict__ Vt,
                                              float* __restrict__ Y) {
  __shared__ _Float16 pbuf[8][16 * 32];
  const int wid = threadIdx.x >> 5, lane = threadIdx.x & 31;
  const int hi = lane >> 4, ln = lane & 15;
  const int bh = blockIdx.y, b = bh >> 4, h = bh & 15;
  const int q0 = (blockIdx.x * 8 + wid) * 16;
  const _Float16* Qb = Qh + (size_t)bh * TSEQ * HD;
  const _Float16* Kb = Kh + (size_t)bh * TSEQ * HD;
  const _Float16* Vb = Vt + (size_t)bh * HD * TSEQ;

  UH16 aq0, aq1;
#pragma unroll
  for (int j = 0; j < 8; ++j) {
    int k0 = (j & 3) * 2 + (j >> 2) * 16 + hi * 8;
    const _Float16* qp = Qb + (size_t)(q0 + ln) * HD + k0;
    aq0.h[2 * j] = qp[0];  aq0.h[2 * j + 1] = qp[1];
    aq1.h[2 * j] = qp[32]; aq1.h[2 * j + 1] = qp[33];
  }

  UF8 yac[4];
  float mrow[8], lrow[8];
#pragma unroll
  for (int j = 0; j < 8; ++j) { mrow[j] = -1e30f; lrow[j] = 0.f; }
#pragma unroll
  for (int c = 0; c < 4; ++c)
#pragma unroll
    for (int j = 0; j < 8; ++j) yac[c].f[j] = 0.f;

  const int nkb = (q0 >> 5) + 1;
  for (int kb = 0; kb < nkb; ++kb) {
    const int kbase = kb * 32;
    UF8 s2[2];
#pragma unroll
    for (int g = 0; g < 2; ++g) {
      UH16 bk0, bk1;
      const _Float16* kp = Kb + (size_t)(kbase + g * 16 + ln) * HD + hi * 8;
#pragma unroll
      for (int e = 0; e < 16; ++e) { bk0.h[e] = kp[e + (e >> 3) * 8]; bk1.h[e] = kp[32 + e + (e >> 3) * 8]; }
      v8f z = {};
      s2[g].v = __builtin_amdgcn_wmma_f32_16x16x32_f16(false, aq0.v, false, bk0.v, (short)0, z, false, false);
      s2[g].v = __builtin_amdgcn_wmma_f32_16x16x32_f16(false, aq1.v, false, bk1.v, (short)0, s2[g].v, false, false);
    }
#pragma unroll
    for (int j = 0; j < 8; ++j) {
      int row = q0 + hi * 8 + j;
      float v0 = (kbase + ln      <= row) ? s2[0].f[j] * SCALEF : -1e30f;
      float v1 = (kbase + 16 + ln <= row) ? s2[1].f[j] * SCALEF : -1e30f;
      float mx = fmaxf(v0, v1);
#pragma unroll
      for (int o = 1; o < 16; o <<= 1) mx = fmaxf(mx, __shfl_xor(mx, o, 16));
      float mn = fmaxf(mrow[j], mx);
      float al = __expf(mrow[j] - mn);
      mrow[j] = mn;
      float p0 = __expf(v0 - mn), p1 = __expf(v1 - mn);
      float rs = p0 + p1;
#pragma unroll
      for (int o = 1; o < 16; o <<= 1) rs += __shfl_xor(rs, o, 16);
      lrow[j] = lrow[j] * al + rs;
#pragma unroll
      for (int c = 0; c < 4; ++c) yac[c].f[j] *= al;
      pbuf[wid][(hi * 8 + j) * 32 + ln]      = (_Float16)p0;
      pbuf[wid][(hi * 8 + j) * 32 + 16 + ln] = (_Float16)p1;
    }
    asm volatile("s_wait_dscnt 0x0" ::: "memory");
    UH16 ap;
#pragma unroll
    for (int j = 0; j < 8; ++j) {
      int k0 = (j & 3) * 2 + (j >> 2) * 16 + hi * 8;
      ap.h[2 * j]     = pbuf[wid][ln * 32 + k0];
      ap.h[2 * j + 1] = pbuf[wid][ln * 32 + k0 + 1];
    }
#pragma unroll
    for (int c = 0; c < 4; ++c) {
      UH16 bv;
      const _Float16* vp = Vb + (size_t)(c * 16 + ln) * TSEQ + kbase + hi * 8;
#pragma unroll
      for (int e = 0; e < 16; ++e) bv.h[e] = vp[e + (e >> 3) * 8];
      yac[c].v = __builtin_amdgcn_wmma_f32_16x16x32_f16(false, ap.v, false, bv.v, (short)0, yac[c].v, false, false);
    }
  }
#pragma unroll
  for (int j = 0; j < 8; ++j) {
    int t = q0 + hi * 8 + j;
    float inv = 1.f / lrow[j];
    float* yrow = Y + ((size_t)b * TSEQ + t) * CDIM + h * HD;
#pragma unroll
    for (int c = 0; c < 4; ++c) yrow[c * 16 + ln] = yac[c].f[j] * inv;
  }
}

extern "C" void kernel_launch(void* const* d_in, const int* in_sizes, int n_in,
                              void* d_out, int out_size, void* d_ws, size_t ws_size,
                              hipStream_t stream) {
  (void)in_sizes; (void)n_in; (void)out_size; (void)ws_size;
  const float* x      = (const float*)d_in[0];
  const float* w_attn = (const float*)d_in[1];
  const float* w_proj = (const float*)d_in[2];
  float* out = (float*)d_out;

  char* ws = (char*)d_ws;
  size_t off = 0;
  auto take = [&](size_t bytes) -> char* {
    char* p = ws + off;
    off += (bytes + 255) & ~(size_t)255;
    return p;
  };

  const int Mtok = BDIM * TSEQ;
  signed char* xq      = (signed char*)take((size_t)Mtok * CDIM);
  float*       invx    = (float*)      take((size_t)Mtok * sizeof(float));
  signed char* wq_attn = (signed char*)take((size_t)3 * CDIM * CDIM);
  float*       scal_a  = (float*)      take(256);
  float*       part_a  = (float*)      take(256 * sizeof(float));
  float*       qkvf    = (float*)      take((size_t)Mtok * 3 * CDIM * sizeof(float));
  _Float16*    Qhm     = (_Float16*)   take((size_t)BDIM * NHEAD * TSEQ * HD * 2);
  _Float16*    Khm     = (_Float16*)   take((size_t)BDIM * NHEAD * TSEQ * HD * 2);
  _Float16*    Vtm     = (_Float16*)   take((size_t)BDIM * NHEAD * TSEQ * HD * 2);
  float*       yf      = (float*)      take((size_t)Mtok * CDIM * sizeof(float));
  signed char* yq      = (signed char*)take((size_t)Mtok * CDIM);
  float*       invy    = (float*)      take((size_t)Mtok * sizeof(float));
  signed char* wq_proj = (signed char*)take((size_t)CDIM * CDIM);
  float*       scal_p  = (float*)      take(256);
  float*       part_p  = (float*)      take(256 * sizeof(float));

  const int n_wa = 3 * CDIM * CDIM, n_wp = CDIM * CDIM;

  k_abs_partial<<<256, 256, 0, stream>>>(w_attn, n_wa, part_a);
  k_abs_final  <<<1,   256, 0, stream>>>(part_a, 256, 1.f / (float)n_wa, scal_a);
  k_quant_w    <<<(n_wa + 255) / 256, 256, 0, stream>>>(w_attn, scal_a, wq_attn, n_wa);
  k_quant_act  <<<Mtok, 256, 0, stream>>>(x, xq, invx);
  k_gemm_iu8   <<<dim3(3 * CDIM / 128, Mtok / 64), 256, 0, stream>>>(xq, wq_attn, invx, scal_a, qkvf, 3 * CDIM);

  k_qkv_post<<<(BDIM * NHEAD * TSEQ * 3) / 8, 256, 0, stream>>>(qkvf, Qhm, Khm, Vtm);
  k_attn    <<<dim3(TSEQ / 16 / 8, BDIM * NHEAD), 256, 0, stream>>>(Qhm, Khm, Vtm, yf);

  k_abs_partial<<<256, 256, 0, stream>>>(w_proj, n_wp, part_p);
  k_abs_final  <<<1,   256, 0, stream>>>(part_p, 256, 1.f / (float)n_wp, scal_p);
  k_quant_w    <<<(n_wp + 255) / 256, 256, 0, stream>>>(w_proj, scal_p, wq_proj, n_wp);
  k_quant_act  <<<Mtok, 256, 0, stream>>>(yf, yq, invy);
  k_gemm_iu8   <<<dim3(CDIM / 128, Mtok / 64), 256, 0, stream>>>(yq, wq_proj, invy, scal_p, out, CDIM);
}
